// BiLSTMCRF_41609643164461
// MI455X (gfx1250) — matrix operand order repaired and checked by reading, not compiled, not run on hardware
//
#include <hip/hip_runtime.h>
#include <hip/hip_bf16.h>
#include <stdint.h>

typedef __attribute__((ext_vector_type(16))) _Float16 v16h;
typedef __attribute__((ext_vector_type(8)))  float    v8f;

#define Bdim 32
#define Tdim 512
#define Ddim 512
#define Ldim 48
#define G4   2048   // 4*D

union Frag16 { v16h v; uint32_t u[8]; uint4 q[2]; _Float16 h[16]; };

__device__ __forceinline__ float sigf(float x) { return 1.0f / (1.0f + __expf(-x)); }

__device__ __forceinline__ v8f wmma_f16(const Frag16& a, const Frag16& b, v8f c) {
  return __builtin_amdgcn_wmma_f32_16x16x32_f16(false, a.v, false, b.v, (short)0, c, false, false);
}

// A-matrix 16x32 f16 fragment: per lane two contiguous 16B chunks (32B apart).
__device__ __forceinline__ void loadA(Frag16& f, const _Float16* rowbase, int k0, int half) {
  const char* p = (const char*)rowbase;
  f.q[0] = *(const uint4*)(p + (size_t)(k0 + half * 8) * 2);
  f.q[1] = *(const uint4*)(p + (size_t)(k0 + 16 + half * 8) * 2);
}

// B-matrix 32x16 f16 fragment: per lane two contiguous 16B chunks (32B apart) of the weight row.
__device__ __forceinline__ void loadB(Frag16& f, const _Float16* rowbase, int k0, int khalf) {
  const char* p = (const char*)rowbase + (size_t)(k0 + khalf * 8) * 2;
  f.q[0] = *(const uint4*)(p);
  f.q[1] = *(const uint4*)(p + 32);
}

// ---------------------------------------------------------------- init sync
__global__ void k_init(int* sync) {
  int i = threadIdx.x;
  if (i < 2 * Tdim) sync[i] = 0;
}

// ------------------------------------------------------- f32 -> f16 convert
__global__ void k_convert(const float* __restrict__ s, _Float16* __restrict__ d, int n) {
  int i = blockIdx.x * blockDim.x + threadIdx.x;
  if (i < n) d[i] = (_Float16)s[i];
}

// ------------------------------------------------------- embedding gather
__global__ void k_embed(const int* __restrict__ tokens, const float* __restrict__ embed,
                        _Float16* __restrict__ emb) {
  int bt = blockIdx.x;                       // one block per (b,t)
  int tok = tokens[bt];
  const float* src = embed + (size_t)tok * Ddim;
  _Float16* dst = emb + (size_t)bt * Ddim;
  for (int i = threadIdx.x; i < Ddim; i += blockDim.x) dst[i] = (_Float16)src[i];
}

// ---------------------------------------------------------- BiLSTM persistent
// grid = 64 blocks: dir = blockIdx.x>>5 (0=fwd,1=bwd), wg = blockIdx.x&31
// each block owns h-features [wg*16, wg*16+16) and the 4 gate rows for them.
// Weight slice (64 rows of W_ih + 64 rows of W_hh, 128 KB f16) is staged in
// dynamic LDS once; the 512-step scan reads B-fragments via ds_load_b128.
__global__ void __launch_bounds__(256)
k_lstm(const _Float16* __restrict__ emb, const _Float16* __restrict__ W4,
       const float* __restrict__ b_ih_f, const float* __restrict__ b_hh_f,
       const float* __restrict__ b_ih_b, const float* __restrict__ b_hh_b,
       const float* __restrict__ init_hidden, const _Float16* __restrict__ hinit16,
       const int* __restrict__ lengths,
       _Float16* __restrict__ hf, _Float16* __restrict__ hbrev, int* __restrict__ sync) {
  extern __shared__ _Float16 Wlds[];   // [2][64][512] f16 = 128 KB (LDS is 320 KB/WGP)
  const int dir  = blockIdx.x >> 5;
  const int wg   = blockIdx.x & 31;
  const int tid  = threadIdx.x;
  const int lane = tid & 31;
  const int w    = tid >> 5;          // 8 waves
  const int mtile = w & 1;            // batches [0,16) or [16,32)
  const int gate  = w >> 1;           // i,f,g,o
  const int half = lane >> 4, r = lane & 15;   // A-frag / C-frag mapping
  const int bn = lane & 15, khalf = lane >> 4; // B-frag mapping

  const _Float16* Wih = W4 + (size_t)dir * 2u * G4 * Ddim;
  const _Float16* Whh = Wih + (size_t)G4 * Ddim;
  const float* bih = dir ? b_ih_b : b_ih_f;
  const float* bhh = dir ? b_hh_b : b_hh_f;
  _Float16* hout = dir ? hbrev : hf;
  const float* init = init_hidden + dir * Ddim;
  const _Float16* init16 = hinit16 + dir * Ddim;   // f16 copy, broadcast row
  int* cnt = sync + dir * Tdim;

  __shared__ float gates_s[4][32][16];
  __shared__ float c_s[32][16];
  __shared__ float bias_s[4][16];

  if (tid < 64) {
    int g = tid >> 4, f = tid & 15;
    int n = g * Ddim + wg * 16 + f;
    bias_s[g][f] = bih[n] + bhh[n];
  }
  for (int cell = tid; cell < 512; cell += 256) {
    int bb = cell >> 4, ff = cell & 15;
    c_s[bb][ff] = init[wg * 16 + ff];   // reference passes h0 as c0 too
  }
  // Stage this WG's weight rows into LDS (one time; amortized over 512 steps).
  {
    uint4* dst = (uint4*)Wlds;
    for (int i = tid; i < 8192; i += 256) {   // 8192 x 16B = 128 KB
      int mat = i >> 12;                      // 0 = W_ih, 1 = W_hh
      int rem = i & 4095;
      int row = rem >> 6;                     // 0..63 local row (gate*16 + f)
      int col = rem & 63;                     // uint4 column within 512-f16 row
      int ng  = (row >> 4) * Ddim + wg * 16 + (row & 15);
      const uint4* src = (const uint4*)((mat ? Whh : Wih) + (size_t)ng * Ddim) + col;
      dst[i] = *src;
    }
  }
  __syncthreads();

  const int brow  = mtile * 16 + r;                    // this lane's batch row
  const int mylen = lengths[brow];
  const _Float16* WihL = Wlds + (size_t)(gate * 16 + bn) * Ddim;          // LDS row
  const _Float16* WhhL = Wlds + (size_t)(64 + gate * 16 + bn) * Ddim;     // LDS row

  for (int t = 0; t < Tdim; ++t) {
    if (t > 0) {
      if (tid == 0)
        while (__hip_atomic_load(&cnt[t - 1], __ATOMIC_ACQUIRE, __HIP_MEMORY_SCOPE_AGENT) < 32)
          __builtin_amdgcn_s_sleep(1);
      __syncthreads();
    }
    int tsrc = t;
    if (dir) { int s2 = mylen - 1 - t; tsrc = (s2 >= 0) ? s2 : t; }  // packed-seq flip
    const _Float16* xrow = emb + ((size_t)brow * Tdim + tsrc) * Ddim;
    // h operand: previous step's output, or the broadcast f16 init row at t==0
    const _Float16* hrow = (t > 0) ? (hout + ((size_t)brow * Tdim + (t - 1)) * Ddim) : init16;

    v8f acc = {};
    for (int kb = 0; kb < 16; ++kb) {
      const int k0 = kb * 32;
      Frag16 ax, ah, bi, bh;
      loadA(ax, xrow, k0, half);
      loadA(ah, hrow, k0, half);
      loadB(bi, WihL, k0, khalf);
      loadB(bh, WhhL, k0, khalf);
      acc = wmma_f16(ax, bi, acc);   // x_t @ W_ih^T
      acc = wmma_f16(ah, bh, acc);   // h_{t-1} @ W_hh^T
    }
#pragma unroll
    for (int g = 0; g < 8; ++g)
      gates_s[gate][mtile * 16 + half * 8 + g][bn] = acc[g];        // C/D layout
    __syncthreads();

    for (int cell = tid; cell < 512; cell += 256) {
      int bb = cell >> 4, ff = cell & 15;
      float ig = sigf(gates_s[0][bb][ff] + bias_s[0][ff]);
      float fg = sigf(gates_s[1][bb][ff] + bias_s[1][ff]);
      float gg = tanhf(gates_s[2][bb][ff] + bias_s[2][ff]);
      float og = sigf(gates_s[3][bb][ff] + bias_s[3][ff]);
      float c = fg * c_s[bb][ff] + ig * gg;
      c_s[bb][ff] = c;
      float h = og * tanhf(c);
      hout[((size_t)bb * Tdim + t) * Ddim + wg * 16 + ff] = (_Float16)h;
    }
    __threadfence();
    __syncthreads();
    if (tid == 0)
      __hip_atomic_fetch_add(&cnt[t], 1, __ATOMIC_RELEASE, __HIP_MEMORY_SCOPE_AGENT);
  }
}

// --------------------------------------------------------- emissions GEMM
// feats[B*T,1024] @ W_emit^T[1024,48] + b_emit, N=48 = 3 WMMA tiles exactly.
__global__ void k_emit(const _Float16* __restrict__ hf, const _Float16* __restrict__ hbrev,
                       const _Float16* __restrict__ Wem, const float* __restrict__ bem,
                       const int* __restrict__ lengths, float* __restrict__ em) {
  const int gx = blockIdx.x;      // M tile: 0..1023
  const int gy = blockIdx.y;      // N tile: 0..2
  const int lane = threadIdx.x;
  const int half = lane >> 4, r = lane & 15;
  const int bn = lane & 15, khalf = lane >> 4;
  const int m = gx * 16 + r;
  const int b = m >> 9, tt = m & 511;
  const int len = lengths[b];
  int s2 = len - 1 - tt;
  const int tsrc = (s2 >= 0) ? s2 : tt;   // un-flip backward hidden states
  const _Float16* frow = hf + ((size_t)b * Tdim + tt) * Ddim;
  const _Float16* brw  = hbrev + ((size_t)b * Tdim + tsrc) * Ddim;
  const int nglob = gy * 16 + bn;
  const _Float16* wrow = Wem + (size_t)nglob * (2 * Ddim);

  v8f acc = {};
  for (int kb = 0; kb < 16; ++kb) {          // K = 0..511  -> forward hidden
    const int k0 = kb * 32;
    Frag16 a, bf;
    loadA(a, frow, k0, half);
    loadB(bf, wrow, k0, khalf);
    acc = wmma_f16(a, bf, acc);
  }
  for (int kb = 0; kb < 16; ++kb) {          // K = 512..1023 -> backward hidden
    const int k0 = kb * 32;
    Frag16 a, bf;
    loadA(a, brw, k0, half);
    loadB(bf, wrow, 512 + k0, khalf);
    acc = wmma_f16(a, bf, acc);
  }
#pragma unroll
  for (int g = 0; g < 8; ++g) {
    int mm = gx * 16 + half * 8 + g;
    em[(size_t)mm * Ldim + nglob] = acc[g] + bem[nglob];
  }
}

// ------------------------------------------------------------- CRF forward
__global__ void k_crf(const float* __restrict__ em, const int* __restrict__ tags,
                      const int* __restrict__ lengths, const float* __restrict__ start_trans,
                      const float* __restrict__ trans, const float* __restrict__ end_trans,
                      float* __restrict__ fwd, float* __restrict__ gold) {
  const int b = blockIdx.x;
  const int tid = threadIdx.x;   // 64 threads
  const int len = lengths[b];
  __shared__ float trans_s[Ldim][Ldim];
  __shared__ float alpha[Ldim];
  for (int i = tid; i < Ldim * Ldim; i += 64) ((float*)trans_s)[i] = trans[i];
  const float* eb = em + (size_t)b * Tdim * Ldim;
  if (tid < Ldim) alpha[tid] = eb[tid] + start_trans[tid];
  __syncthreads();
  for (int t = 1; t < Tdim; ++t) {
    float nv = 0.0f;
    if (tid < Ldim) {
      if (t < len) {
        float mx = -1e30f;
        for (int k = 0; k < Ldim; ++k) { float v = alpha[k] + trans_s[tid][k]; mx = fmaxf(mx, v); }
        float s = 0.0f;
        for (int k = 0; k < Ldim; ++k) s += __expf(alpha[k] + trans_s[tid][k] - mx);
        nv = eb[t * Ldim + tid] + mx + __logf(s);
      } else {
        nv = alpha[tid];
      }
    }
    __syncthreads();
    if (tid < Ldim) alpha[tid] = nv;
    __syncthreads();
  }
  if (tid == 0) {
    float mx = -1e30f;
    for (int j = 0; j < Ldim; ++j) mx = fmaxf(mx, alpha[j] + end_trans[j]);
    float s = 0.0f;
    for (int j = 0; j < Ldim; ++j) s += __expf(alpha[j] + end_trans[j] - mx);
    fwd[b] = mx + __logf(s);
    // gold path score (serial, deterministic)
    const int* tg = tags + (size_t)b * Tdim;
    float g = start_trans[tg[0]] + eb[tg[0]];
    for (int t = 1; t < Tdim; ++t)
      if (t < len) g += trans[tg[t] * Ldim + tg[t - 1]] + eb[t * Ldim + tg[t]];
    g += end_trans[tg[len - 1]];
    gold[b] = g;
  }
}

// --------------------------------------------------------------- final sum
__global__ void k_final(const float* __restrict__ fwd, const float* __restrict__ gold,
                        float* __restrict__ out) {
  float s = 0.0f;
  for (int b = 0; b < Bdim; ++b) s += fwd[b] - gold[b];
  out[0] = s;
}

// ----------------------------------------------------------------- launcher
extern "C" void kernel_launch(void* const* d_in, const int* in_sizes, int n_in,
                              void* d_out, int out_size, void* d_ws, size_t ws_size,
                              hipStream_t stream) {
  const int*   tokens  = (const int*)d_in[0];
  const int*   tags    = (const int*)d_in[1];
  const int*   lengths = (const int*)d_in[2];
  const float* embed   = (const float*)d_in[3];
  const float* Wih_f   = (const float*)d_in[4];
  const float* Whh_f   = (const float*)d_in[5];
  const float* bih_f   = (const float*)d_in[6];
  const float* bhh_f   = (const float*)d_in[7];
  const float* Wih_b   = (const float*)d_in[8];
  const float* Whh_b   = (const float*)d_in[9];
  const float* bih_b   = (const float*)d_in[10];
  const float* bhh_b   = (const float*)d_in[11];
  const float* inith   = (const float*)d_in[12];
  const float* Wemit   = (const float*)d_in[13];
  const float* bemit   = (const float*)d_in[14];
  const float* strans  = (const float*)d_in[15];
  const float* trans   = (const float*)d_in[16];
  const float* etrans  = (const float*)d_in[17];
  float* out = (float*)d_out;

  char* ws = (char*)d_ws;
  constexpr size_t SZ_EMB = (size_t)Bdim * Tdim * Ddim * 2;   // 16 MB
  constexpr size_t SZ_H   = SZ_EMB;
  constexpr size_t SZ_W   = (size_t)G4 * Ddim * 2;            // 2 MB per matrix
  constexpr size_t SZ_WE  = (size_t)Ldim * 2 * Ddim * 2;
  constexpr size_t SZ_EM  = (size_t)Bdim * Tdim * Ldim * 4;
  _Float16* emb    = (_Float16*)(ws);
  _Float16* hf     = (_Float16*)(ws + SZ_EMB);
  _Float16* hbrev  = (_Float16*)(ws + SZ_EMB + SZ_H);
  _Float16* W4     = (_Float16*)(ws + SZ_EMB + 2 * SZ_H);
  _Float16* WemH   = (_Float16*)(ws + SZ_EMB + 2 * SZ_H + 4 * SZ_W);
  float*    emis   = (float*)   (ws + SZ_EMB + 2 * SZ_H + 4 * SZ_W + SZ_WE);
  float*    fwd    = (float*)   (ws + SZ_EMB + 2 * SZ_H + 4 * SZ_W + SZ_WE + SZ_EM);
  float*    gold   = fwd + 32;
  int*      sync   = (int*)(fwd + 64);
  _Float16* hinit16 = (_Float16*)(sync + 2 * Tdim);           // 2*D f16

  k_init<<<1, 1024, 0, stream>>>(sync);
  k_embed<<<Bdim * Tdim, 256, 0, stream>>>(tokens, embed, emb);

  const int nW = G4 * Ddim;
  k_convert<<<(nW + 255) / 256, 256, 0, stream>>>(Wih_f, W4 + 0 * (size_t)nW, nW);
  k_convert<<<(nW + 255) / 256, 256, 0, stream>>>(Whh_f, W4 + 1 * (size_t)nW, nW);
  k_convert<<<(nW + 255) / 256, 256, 0, stream>>>(Wih_b, W4 + 2 * (size_t)nW, nW);
  k_convert<<<(nW + 255) / 256, 256, 0, stream>>>(Whh_b, W4 + 3 * (size_t)nW, nW);
  const int nWe = Ldim * 2 * Ddim;
  k_convert<<<(nWe + 255) / 256, 256, 0, stream>>>(Wemit, WemH, nWe);
  k_convert<<<4, 256, 0, stream>>>(inith, hinit16, 2 * Ddim);  // f16 init rows

  k_lstm<<<64, 256, 131072, stream>>>(emb, W4, bih_f, bhh_f, bih_b, bhh_b, inith, hinit16,
                                      lengths, hf, hbrev, sync);
  k_emit<<<dim3(Bdim * Tdim / 16, 3), 32, 0, stream>>>(hf, hbrev, WemH, bemit, lengths, emis);
  k_crf<<<Bdim, 64, 0, stream>>>(emis, tags, lengths, strans, trans, etrans, fwd, gold);
  k_final<<<1, 1, 0, stream>>>(fwd, gold, out);

  (void)in_sizes; (void)n_in; (void)out_size; (void)ws_size;
}
